// AnisotropicGNNLayer_13554916786282
// MI455X (gfx1250) — matrix operand order repaired and checked by reading, not compiled, not run on hardware
//
#include <hip/hip_runtime.h>
#include <hip/hip_bf16.h>
#include <math.h>

typedef __attribute__((ext_vector_type(16))) _Float16 v16h;
typedef __attribute__((ext_vector_type(8)))  float    v8f;

#define BDIM   16
#define FRDIM  512
#define JDIM   53
#define EDIM   52
#define DIN    128
#define DOUT   256
#define MROWS  (BDIM * FRDIM)   // 8192 collapsed (b, fr) rows
#define MTILE  128              // rows per block (8 waves x 16)
#define LN_EPS 1e-5f

// Index (in halves) into the LDS-packed WMMA B-fragment layout.
// B tile (kb, nb) is 32x16 f16 stored as: lane = ni + (ki & 16),
// within-lane half index = ki & 15 (VGPR i holds K pair {2i, 2i+1}).
__device__ __forceinline__ int fragIdx(int k, int n) {
    int kb = k >> 5, ki = k & 31;
    int nb = n >> 4, ni = n & 15;
    int lane = ni + (ki & 16);
    int kk = ki & 15;
    return ((kb << 4) + nb) * 512 + lane * 16 + kk;
}

__global__ __launch_bounds__(256)
void gnn_fused_kernel(const float* __restrict__ f,
                      const float* __restrict__ W,
                      const float* __restrict__ pose,
                      const float* __restrict__ gamma,
                      const float* __restrict__ beta,
                      const float* __restrict__ resW,
                      float* __restrict__ out)
{
    __shared__ __align__(32) _Float16 lds[DIN * DOUT]; // 64 KB fragment buffer

    const int j    = blockIdx.y;          // joint (uniform per block)
    const int m0   = blockIdx.x * MTILE;  // first (b,fr) row of this block
    const int tid  = threadIdx.x;
    const int lane = tid & 31;
    const int wave = tid >> 5;
    const int nl   = lane & 15;
    const int hi   = lane >> 4;           // 0: lanes 0-15, 1: lanes 16-31

    // ---- pack W[j] (fp32 row-major 128x256) into f16 WMMA-B fragments ----
    if (j < EDIM) {
        const float* Wj = W + (size_t)j * DIN * DOUT;
        const int n = tid; // one output column per thread -> coalesced reads
#pragma unroll 4
        for (int k = 0; k < DIN; k += 2) {
            int h = fragIdx(k, n);
            lds[h]     = (_Float16)Wj[(size_t)k * DOUT + n];
            lds[h + 1] = (_Float16)Wj[(size_t)(k + 1) * DOUT + n];
        }
    }
    __syncthreads();

    const int rowA = m0 + wave * 16 + nl;  // A-matrix row held by this lane
    const float* frow_j  = f + ((size_t)rowA * JDIM + j) * DIN;
    const float* frow_j1 = frow_j + DIN;   // joint j+1 (only used when j < 52)
    __builtin_prefetch(frow_j, 0, 1);
    __builtin_prefetch(frow_j + DIN, 0, 1);

    v8f acc[16] = {};  // 16 rows x 256 cols fp32 accumulator (16 C tiles)

    // ---- message GEMM: (f[:,j+1]-f[:,j]) @ W[j], K = 128 in 4 WMMA steps ----
    if (j < EDIM) {
#pragma unroll
        for (int kb = 0; kb < 4; ++kb) {
            const int c0 = kb * 32 + hi * 16;  // contiguous k = 16*hi + s, matching B fragments
            const int c1 = c0 + 8;
            v16h a;
#pragma unroll
            for (int t = 0; t < 8; ++t)
                a[t]     = (_Float16)(frow_j1[c0 + t] - frow_j[c0 + t]);
#pragma unroll
            for (int t = 0; t < 8; ++t)
                a[8 + t] = (_Float16)(frow_j1[c1 + t] - frow_j[c1 + t]);
#pragma unroll
            for (int nb = 0; nb < 16; ++nb) {
                v16h bf = *(const v16h*)(lds + ((kb << 4) + nb) * 512 + lane * 16);
                acc[nb] = __builtin_amdgcn_wmma_f32_16x16x32_f16(
                    false, a, false, bf, (short)0, acc[nb], false, false);
            }
        }
    }

    // ---- epilogue params: pose_emb[j,:], gamma, beta for my 16 columns ----
    float pe[16], ga[16], be[16];
#pragma unroll
    for (int nb = 0; nb < 16; ++nb) {
        int n = nb * 16 + nl;
        pe[nb] = pose[j * DOUT + n];
        ga[nb] = gamma[n];
        be[nb] = beta[n];
    }

    // ---- +pose_emb, LayerNorm over DOUT, exact GELU (in registers) ----
    // C layout: reg r holds row r (lanes 0-15) / row r+8 (lanes 16-31),
    // so xor-shuffles with masks < 16 reduce exactly one row per half-wave.
#pragma unroll
    for (int r = 0; r < 8; ++r) {
        float s = 0.0f;
#pragma unroll
        for (int nb = 0; nb < 16; ++nb) {
            float x = acc[nb][r] + pe[nb];
            acc[nb][r] = x;
            s += x;
        }
        s += __shfl_xor(s, 1, 32);
        s += __shfl_xor(s, 2, 32);
        s += __shfl_xor(s, 4, 32);
        s += __shfl_xor(s, 8, 32);
        float mu = s * (1.0f / DOUT);
        float ss = 0.0f;
#pragma unroll
        for (int nb = 0; nb < 16; ++nb) {
            float d = acc[nb][r] - mu;
            ss += d * d;
        }
        ss += __shfl_xor(ss, 1, 32);
        ss += __shfl_xor(ss, 2, 32);
        ss += __shfl_xor(ss, 4, 32);
        ss += __shfl_xor(ss, 8, 32);
        float rstd = rsqrtf(ss * (1.0f / DOUT) + LN_EPS);
#pragma unroll
        for (int nb = 0; nb < 16; ++nb) {
            float x = (acc[nb][r] - mu) * rstd * ga[nb] + be[nb];
            acc[nb][r] = 0.5f * x * (1.0f + erff(x * 0.70710678118654752f));
        }
    }

    // ---- repack res_W^T into the same LDS buffer: B[k][n] = res_W[n][k] ----
    __syncthreads();   // all waves done reading W fragments
    {
        const int n = tid;
        const float* rn = resW + (size_t)n * DIN;
#pragma unroll 4
        for (int k = 0; k < DIN; k += 2) {
            int h = fragIdx(k, n);
            lds[h]     = (_Float16)rn[k];
            lds[h + 1] = (_Float16)rn[k + 1];
        }
    }
    __syncthreads();

    // ---- residual GEMM f @ res_W^T, accumulated onto post-GELU values ----
#pragma unroll
    for (int kb = 0; kb < 4; ++kb) {
        const int c0 = kb * 32 + hi * 16;
        const int c1 = c0 + 8;
        v16h a;
#pragma unroll
        for (int t = 0; t < 8; ++t) a[t]     = (_Float16)frow_j[c0 + t];
#pragma unroll
        for (int t = 0; t < 8; ++t) a[8 + t] = (_Float16)frow_j[c1 + t];
#pragma unroll
        for (int nb = 0; nb < 16; ++nb) {
            v16h bf = *(const v16h*)(lds + ((kb << 4) + nb) * 512 + lane * 16);
            acc[nb] = __builtin_amdgcn_wmma_f32_16x16x32_f16(
                false, a, false, bf, (short)0, acc[nb], false, false);
        }
    }

    // ---- store 16x256 fp32 tile ----
#pragma unroll
    for (int r = 0; r < 8; ++r) {
        size_t gr = (size_t)(m0 + wave * 16 + r + (hi << 3));
        float* orow = out + (gr * JDIM + j) * DOUT + nl;
#pragma unroll
        for (int nb = 0; nb < 16; ++nb) orow[nb * 16] = acc[nb][r];
    }
}

extern "C" void kernel_launch(void* const* d_in, const int* in_sizes, int n_in,
                              void* d_out, int out_size, void* d_ws, size_t ws_size,
                              hipStream_t stream) {
    (void)in_sizes; (void)n_in; (void)out_size; (void)d_ws; (void)ws_size;
    const float* f     = (const float*)d_in[0];
    const float* W     = (const float*)d_in[1];
    const float* pose  = (const float*)d_in[2];
    const float* gamma = (const float*)d_in[3];
    const float* beta  = (const float*)d_in[4];
    const float* resW  = (const float*)d_in[5];
    // d_in[6], d_in[7] (src/dst) encode the fixed chain e -> (e+1, e); folded in.
    dim3 grid(MROWS / MTILE, JDIM);
    gnn_fused_kernel<<<grid, dim3(256), 0, stream>>>(
        f, W, pose, gamma, beta, resW, (float*)d_out);
}
